// NeuralNetwork_49804440764428
// MI455X (gfx1250) — matrix operand order repaired and checked by reading, not compiled, not run on hardware
//
#include <hip/hip_runtime.h>
#include <hip/hip_bf16.h>

typedef __attribute__((ext_vector_type(16))) _Float16 v16h;
typedef __attribute__((ext_vector_type(8)))  float    v8f;
typedef __attribute__((ext_vector_type(2)))  float    v2f;

#define LSTM_B 2048
#define LSTM_T 512
#define LSTM_I 4
#define LSTM_H 64
#define LSTM_G 256   // 4*H gate columns (i,f,g,o)
#define LSTM_O 4
#define MTILE  16    // batch rows per block
#define THREADS 256  // 8 waves (wave32)

// fp32 K=4 WMMA for the x-contribution (probe-confirmed to lower on gfx1250)
#if defined(__has_builtin)
#if __has_builtin(__builtin_amdgcn_wmma_f32_16x16x4_f32)
#define USE_WMMA_F32X4 1
#endif
#if __has_builtin(__builtin_amdgcn_tanhf)
#define USE_HW_TANH 1
#endif
#endif
#ifndef USE_WMMA_F32X4
#define USE_WMMA_F32X4 0
#endif
#ifndef USE_HW_TANH
#define USE_HW_TANH 0
#endif

// fast tanh: single v_tanh_f32 on gfx1250, else exp + raw v_rcp_f32
__device__ __forceinline__ float tanh_fast(float x) {
#if USE_HW_TANH
  return __builtin_amdgcn_tanhf(x);
#else
  float e = __expf(-2.0f * fabsf(x));           // e in (0,1]; no overflow
  float r = (1.0f - e) * __builtin_amdgcn_rcpf(1.0f + e);
  return copysignf(r, x);
#endif
}
// sigmoid(x) = 0.5*tanh(x/2) + 0.5 : mul + tanh + fma
__device__ __forceinline__ float sigmoid_fast(float x) {
  return fmaf(0.5f, tanh_fast(0.5f * x), 0.5f);
}

__global__ __launch_bounds__(THREADS, 2)
void lstm_fused_wmma(const float* __restrict__ x,
                     const float* __restrict__ W_ih,
                     const float* __restrict__ W_hh,
                     const float* __restrict__ b_ih,
                     const float* __restrict__ b_hh,
                     const float* __restrict__ W_fc,
                     const float* __restrict__ b_fc,
                     float* __restrict__ out) {
  __shared__ __align__(16) float     gates[MTILE][LSTM_G];  // 16 KB pre-activations
  __shared__ __align__(32) _Float16  hswz[2][32][16];       // 2 KB h in WMMA A-frag layout
  __shared__ __align__(16) float     h32[MTILE][LSTM_H];    // 4 KB final-step h (fp32)
  __shared__ __align__(16) float     xs[MTILE][LSTM_I];     // 256 B staged x_t

  const int tid  = threadIdx.x;
  const int wave = tid >> 5;
  const int lane = tid & 31;
  const int l16  = lane & 15;
  const int g    = lane >> 4;          // lane group (0: lanes 0-15, 1: 16-31)
  const int rowb = blockIdx.x * MTILE;

  // ---------------- one-time setup: resident weight fragments ----------------
  const int ncol0 = wave * 32;         // this wave's 32 gate columns
  v16h  bw[2][2];                      // W_hh^T B-frags: [n-tile][k-chunk]
  float biasv[2];
  float wi4[2][4];                     // W_ih rows (VALU fallback path)
#if USE_WMMA_F32X4
  v2f   bx[2];                         // W_ih^T B-frags for 16x16x4 f32 WMMA
#endif
#pragma unroll
  for (int nt = 0; nt < 2; ++nt) {
    const int n = ncol0 + nt * 16 + l16;       // global gate column
    biasv[nt] = b_ih[n] + b_hh[n];
    const float* wrow = W_hh + n * LSTM_H;     // W_hh[n][k] == B[k][n]
#pragma unroll
    for (int j = 0; j < 16; ++j) {
      // B-frag (16-bit, 32x16): lane group g holds K = kc*32 + (j>>3)*16 + g*8 + (j&7)
      bw[nt][0][j] = (_Float16)wrow[(j >> 3) * 16 + g * 8 + (j & 7)];
      bw[nt][1][j] = (_Float16)wrow[32 + (j >> 3) * 16 + g * 8 + (j & 7)];
    }
#pragma unroll
    for (int k = 0; k < 4; ++k) wi4[nt][k] = W_ih[n * LSTM_I + k];
#if USE_WMMA_F32X4
    // f32 B 4x16: lane group g holds K = 2g, 2g+1
    bx[nt].x = W_ih[n * LSTM_I + 2 * g];
    bx[nt].y = W_ih[n * LSTM_I + 2 * g + 1];
#endif
  }

  // zero initial h fragment, stage x_0 (x is streamed once -> non-temporal)
  for (int i = tid; i < 2 * 32 * 16; i += THREADS)
    ((_Float16*)hswz)[i] = (_Float16)0.0f;
  if (tid < MTILE * LSTM_I) {
    const int m = tid >> 2, k = tid & 3;
    xs[m][k] =
        __builtin_nontemporal_load(&x[((size_t)(rowb + m) * LSTM_T + 0) * LSTM_I + k]);
  }
  float creg[4] = {0.0f, 0.0f, 0.0f, 0.0f};    // cell state, 4 elems/thread
  __syncthreads();

  // ---------------- recurrence over T ----------------
  for (int t = 0; t < LSTM_T; ++t) {
    // ---- gate phase: gates = bias + x_t@W_ih^T + h@W_hh^T ----
    v16h ah0 = *(const v16h*)&hswz[0][lane][0];
    v16h ah1 = *(const v16h*)&hswz[1][lane][0];
#if USE_WMMA_F32X4
    v2f ax;                                     // f32 A 16x4: K = 2g, 2g+1
    ax.x = xs[l16][2 * g];
    ax.y = xs[l16][2 * g + 1];
#endif
#pragma unroll
    for (int nt = 0; nt < 2; ++nt) {
      v8f acc;
#if USE_WMMA_F32X4
      v8f cb;
#pragma unroll
      for (int r = 0; r < 8; ++r) cb[r] = biasv[nt];
      acc = __builtin_amdgcn_wmma_f32_16x16x4_f32(
          false, ax, false, bx[nt], (short)0, cb, false, false);
#else
#pragma unroll
      for (int r = 0; r < 8; ++r) {
        const int m = r + g * 8;
        acc[r] = biasv[nt] + xs[m][0] * wi4[nt][0] + xs[m][1] * wi4[nt][1]
                           + xs[m][2] * wi4[nt][2] + xs[m][3] * wi4[nt][3];
      }
#endif
      acc = __builtin_amdgcn_wmma_f32_16x16x32_f16(
          false, ah0, false, bw[nt][0], (short)0, acc, false, false);
      acc = __builtin_amdgcn_wmma_f32_16x16x32_f16(
          false, ah1, false, bw[nt][1], (short)0, acc, false, false);
      // D layout: VGPR r, lane group g -> row M = r + 8g, col N = l16
#pragma unroll
      for (int r = 0; r < 8; ++r)
        gates[r + g * 8][ncol0 + nt * 16 + l16] = acc[r];
    }
    __syncthreads();

    // ---- elementwise LSTM update; c resident in registers ----
#pragma unroll
    for (int q = 0; q < 4; ++q) {
      const int e = tid + q * THREADS;          // 0..1023
      const int m = e >> 6, k = e & 63;
      const float gi = gates[m][k];
      const float gf = gates[m][64 + k];
      const float gc = gates[m][128 + k];
      const float go = gates[m][192 + k];
      const float c  = sigmoid_fast(gf) * creg[q] + sigmoid_fast(gi) * tanh_fast(gc);
      creg[q] = c;
      const float h = sigmoid_fast(go) * tanh_fast(c);
      // scatter h into the 16-bit A-fragment (16x32) swizzled layout
      const int kc = k >> 5, kk = k & 31;
      const int gw = (kk >> 3) & 1;             // lane group
      const int j  = (kk >> 4) * 8 + (kk & 7);  // element within v16h
      hswz[kc][gw * 16 + m][j] = (_Float16)h;
      if (t == LSTM_T - 1) h32[m][k] = h;
    }
    // stage x_{t+1}
    if (tid < MTILE * LSTM_I && t + 1 < LSTM_T) {
      const int m = tid >> 2, k = tid & 3;
      xs[m][k] = __builtin_nontemporal_load(
          &x[((size_t)(rowb + m) * LSTM_T + (t + 1)) * LSTM_I + k]);
    }
    if (lane == 0 && t + 8 < LSTM_T)
      __builtin_prefetch(&x[((size_t)rowb * LSTM_T + (t + 8)) * LSTM_I], 0, 1);
    __syncthreads();
  }

  // ---------------- tiny FC head: out = h @ W_fc^T + b_fc ----------------
  if (tid < MTILE * LSTM_O) {
    const int m = tid >> 2, o = tid & 3;
    float s = b_fc[o];
    const float* wf = W_fc + o * LSTM_H;
#pragma unroll
    for (int k = 0; k < LSTM_H; ++k) s += h32[m][k] * wf[k];
    out[(rowb + m) * LSTM_O + o] = s;
  }
}

extern "C" void kernel_launch(void* const* d_in, const int* in_sizes, int n_in,
                              void* d_out, int out_size, void* d_ws, size_t ws_size,
                              hipStream_t stream) {
  const float* x    = (const float*)d_in[0];
  const float* W_ih = (const float*)d_in[1];
  const float* W_hh = (const float*)d_in[2];
  const float* b_ih = (const float*)d_in[3];
  const float* b_hh = (const float*)d_in[4];
  const float* W_fc = (const float*)d_in[5];
  const float* b_fc = (const float*)d_in[6];
  float* out = (float*)d_out;

  dim3 grid(LSTM_B / MTILE);   // 128 independent batch tiles
  dim3 block(THREADS);
  lstm_fused_wmma<<<grid, block, 0, stream>>>(x, W_ih, W_hh, b_ih, b_hh,
                                              W_fc, b_fc, out);
}
